// FeatureFusionByGNN_34857954574437
// MI455X (gfx1250) — matrix operand order repaired and checked by reading, not compiled, not run on hardware
//
#include <hip/hip_runtime.h>
#include <hip/hip_bf16.h>

// ---------------- problem constants (match reference) ----------------
#define NN   50000
#define GG   64
#define HH   8
#define CC   32
#define HC   256          // hidden = H*C
#define INC  256          // in channels
#define E1   250000
#define E2   500000       // 2E concatenated edges
#define TBL_BITS 21
#define TBL_SIZE (1u << TBL_BITS)

// ---------------- WMMA vector types (CDNA5 wave32) ----------------
typedef __attribute__((ext_vector_type(16))) __bf16 v16bf;
typedef __attribute__((ext_vector_type(8)))  float  v8f;

union ABFrag {            // two aligned 16B chunks <-> one bf16x16 fragment
    uint4  u[2];
    v16bf  v;
};

__device__ __forceinline__ unsigned short f32_to_bf16_rne(float f) {
    unsigned u = __float_as_uint(f);
    u += 0x7FFFu + ((u >> 16) & 1u);
    return (unsigned short)(u >> 16);
}

// monotonic float<->uint encoding for atomicMax over signed floats
__device__ __forceinline__ unsigned enc_f32(float f) {
    unsigned u = __float_as_uint(f);
    return (u & 0x80000000u) ? ~u : (u | 0x80000000u);
}
__device__ __forceinline__ float dec_f32(unsigned e) {
    unsigned u = (e & 0x80000000u) ? (e & 0x7FFFFFFFu) : ~e;
    return __uint_as_float(u);
}

// ---------------- kernels ----------------

// x = concat(feat1, feat2) -> bf16 [N, 256]
__global__ void build_xh_kernel(const float* __restrict__ f1,
                                const float* __restrict__ f2,
                                unsigned short* __restrict__ xh) {
    int gid = blockIdx.x * blockDim.x + threadIdx.x;
    if (gid >= NN * HC) return;
    int n = gid >> 8, c = gid & 255;
    float f = (c < 128) ? f1[n * 128 + c] : f2[n * 128 + (c - 128)];
    xh[gid] = f32_to_bf16_rne(f);
}

__global__ void f32_to_bf16_kernel(const float* __restrict__ src,
                                   unsigned short* __restrict__ dst, int n) {
    int gid = blockIdx.x * blockDim.x + threadIdx.x;
    if (gid < n) dst[gid] = f32_to_bf16_rne(src[gid]);
}

// Out[N,256] = Xh[N,256] @ Wh[256,256]^T + bias  via v_wmma_f32_16x16x32_bf16.
// Register-blocked: block = 128 threads = 4 waves; wave w owns a 16(M) x 64(N)
// tile (N-tiles 4w..4w+3) of M-tile blockIdx.x. A fragment loaded ONCE per
// K-step and reused across 4 independent WMMA accumulator chains (XDL stays
// full, 4x less A traffic). grid = (3125, 1).
// A fragment (16x32 bf16, ISA 7.12.2): lane = M row (lane&15); lanes<16 hold
// K 0..15, lanes>=16 hold K 16..31 (contiguous, not ISA table order) -> two 16B chunks per lane.
// B fragment (32x16 bf16): lane = N col (lane&15); lanes<16 hold K 0..15,
// lanes>=16 K 16..31.
// D (16x16 f32, v8f): lane&15 = N col, element j = M row j + 8*(lane>=16).
__global__ void gemm_wmma_bf16_kernel(const unsigned short* __restrict__ Xh,
                                      const unsigned short* __restrict__ Wh,
                                      const float* __restrict__ bias,
                                      float* __restrict__ Out) {
    const int wave   = threadIdx.x >> 5;     // 0..3
    const int lane   = threadIdx.x & 31;
    const int half   = lane >> 4;            // 0 or 1
    const int l15    = lane & 15;
    const int mtile  = blockIdx.x;
    const int ntile0 = wave * 4;             // first of 4 N-tiles for this wave
    const int row    = mtile * 16 + l15;     // A row this lane feeds

    const unsigned short* arow = Xh + (size_t)row * INC;
    // B rows for the 4 output-column groups this lane feeds
    const unsigned short* brow = Wh + (size_t)(ntile0 * 16 + l15) * INC;

    ABFrag a, b0, b1, b2, b3;
    v8f acc0 = {}, acc1 = {}, acc2 = {}, acc3 = {};
#pragma unroll
    for (int ks = 0; ks < INC; ks += 32) {
        __builtin_prefetch(arow + ks + 64, 0, 1);   // global_prefetch_b8
        a.u[0]  = *(const uint4*)(arow + ks + half * 16);       // K ks+{0..7}/{16..23}
        a.u[1]  = *(const uint4*)(arow + ks + half * 16 + 8);   // K ks+{8..15}/{24..31}
        const int bo = ks + half * 16;
        b0.u[0] = *(const uint4*)(brow + 0 * 16 * INC + bo);
        b0.u[1] = *(const uint4*)(brow + 0 * 16 * INC + bo + 8);
        b1.u[0] = *(const uint4*)(brow + 1 * 16 * INC + bo);
        b1.u[1] = *(const uint4*)(brow + 1 * 16 * INC + bo + 8);
        b2.u[0] = *(const uint4*)(brow + 2 * 16 * INC + bo);
        b2.u[1] = *(const uint4*)(brow + 2 * 16 * INC + bo + 8);
        b3.u[0] = *(const uint4*)(brow + 3 * 16 * INC + bo);
        b3.u[1] = *(const uint4*)(brow + 3 * 16 * INC + bo + 8);
        acc0 = __builtin_amdgcn_wmma_f32_16x16x32_bf16(
                   false, a.v, false, b0.v, (short)0, acc0, false, false);
        acc1 = __builtin_amdgcn_wmma_f32_16x16x32_bf16(
                   false, a.v, false, b1.v, (short)0, acc1, false, false);
        acc2 = __builtin_amdgcn_wmma_f32_16x16x32_bf16(
                   false, a.v, false, b2.v, (short)0, acc2, false, false);
        acc3 = __builtin_amdgcn_wmma_f32_16x16x32_bf16(
                   false, a.v, false, b3.v, (short)0, acc3, false, false);
    }

    float* obase = Out + (size_t)(mtile * 16) * HC;
#pragma unroll
    for (int t = 0; t < 4; ++t) {
        const v8f acc = (t == 0) ? acc0 : (t == 1) ? acc1 : (t == 2) ? acc2 : acc3;
        const int col = (ntile0 + t) * 16 + l15;
        const float bb = bias[col];
#pragma unroll
        for (int j = 0; j < 8; ++j) {
            int m = j + half * 8;
            obase[(size_t)m * HC + col] = acc[j] + bb;
        }
    }
}

// first-occurrence dedup of concatenated edge list via atomicCAS hash set
__global__ void dedup_kernel(const int* __restrict__ ei1, const int* __restrict__ ei2,
                             unsigned long long* __restrict__ table,
                             int* __restrict__ keep,
                             int* __restrict__ se, int* __restrict__ de) {
    int e = blockIdx.x * blockDim.x + threadIdx.x;
    if (e >= E2) return;
    int s, d;
    if (e < E1) { s = ei1[e];      d = ei1[E1 + e]; }
    else        { int j = e - E1; s = ei2[j]; d = ei2[E1 + j]; }
    se[e] = s; de[e] = d;
    unsigned long long code = (unsigned long long)(unsigned)s * (unsigned long long)NN
                            + (unsigned long long)(unsigned)d;
    unsigned long long h = (code * 0x9E3779B97F4A7C15ull) >> (64 - TBL_BITS);
    int k = 0;
    for (;;) {
        unsigned long long old = atomicCAS(&table[h], ~0ull, code);
        if (old == ~0ull) { k = 1; break; }     // first occurrence
        if (old == code)  { k = 0; break; }     // duplicate
        h = (h + 1) & (unsigned long long)(TBL_SIZE - 1);
    }
    keep[e] = k;
}

// per-(edge, head): score = <q[dst], k[src]>/sqrt(C); segment max via encoded atomicMax
__global__ void scores_kernel(const float* __restrict__ q, const float* __restrict__ k,
                              const int* __restrict__ se, const int* __restrict__ de,
                              const int* __restrict__ keep,
                              float* __restrict__ sc, unsigned* __restrict__ menc) {
    int gid = blockIdx.x * blockDim.x + threadIdx.x;
    if (gid >= E2 * HH) return;
    int e = gid >> 3, h = gid & 7;
    if (!keep[e]) { sc[gid] = 0.0f; return; }
    int s = se[e], d = de[e];
    const float* qp = q + (size_t)d * HC + h * CC;
    const float* kp = k + (size_t)s * HC + h * CC;
    float acc = 0.0f;
#pragma unroll
    for (int c = 0; c < CC; c += 4) {
        float4 av = *(const float4*)(qp + c);
        float4 bv = *(const float4*)(kp + c);
        acc += av.x * bv.x + av.y * bv.y + av.z * bv.z + av.w * bv.w;
    }
    float scv = acc * 0.17677669529663689f;     // 1/sqrt(32)
    sc[gid] = scv;
    atomicMax(&menc[(size_t)d * HH + h], enc_f32(scv));
}

// e = exp(score - m[dst]); denom = segment_sum(e)
__global__ void expsum_kernel(const int* __restrict__ de, const int* __restrict__ keep,
                              const unsigned* __restrict__ menc,
                              float* __restrict__ sc, float* __restrict__ denom) {
    int gid = blockIdx.x * blockDim.x + threadIdx.x;
    if (gid >= E2 * HH) return;
    int e = gid >> 3, h = gid & 7;
    if (!keep[e]) return;
    int d = de[e];
    float m  = dec_f32(menc[(size_t)d * HH + h]);
    float ev = __expf(sc[gid] - m);
    sc[gid]  = ev;
    atomicAdd(&denom[(size_t)d * HH + h], ev);
}

// agg[dst] += alpha * v[src]
__global__ void aggregate_kernel(const int* __restrict__ se, const int* __restrict__ de,
                                 const int* __restrict__ keep,
                                 const float* __restrict__ sc, const float* __restrict__ denom,
                                 const float* __restrict__ v, float* __restrict__ agg) {
    int gid = blockIdx.x * blockDim.x + threadIdx.x;
    if (gid >= E2 * HH) return;
    int e = gid >> 3, h = gid & 7;
    if (!keep[e]) return;
    int s = se[e], d = de[e];
    float alpha = sc[gid] / fmaxf(denom[(size_t)d * HH + h], 1e-16f);
    const float* vp = v + (size_t)s * HC + h * CC;
    float* ap = agg + (size_t)d * HC + h * CC;
#pragma unroll
    for (int c = 0; c < CC; ++c) atomicAdd(&ap[c], alpha * vp[c]);
}

// out = agg + skip; accumulate per-graph sum / sumsq / count
__global__ void node_stats_kernel(const float* __restrict__ sout, float* __restrict__ out,
                                  const int* __restrict__ batch,
                                  float* __restrict__ gsum, float* __restrict__ gsumsq,
                                  float* __restrict__ gcnt) {
    int n = blockIdx.x * blockDim.x + threadIdx.x;
    if (n >= NN) return;
    int g = batch[n];
    float s1 = 0.0f, s2 = 0.0f;
    float* op = out + (size_t)n * HC;
    const float* sp = sout + (size_t)n * HC;
#pragma unroll 8
    for (int c = 0; c < HC; ++c) {
        float v = op[c] + sp[c];
        op[c] = v;
        s1 += v;
        s2 += v * v;
    }
    atomicAdd(&gsum[g], s1);
    atomicAdd(&gsumsq[g], s2);
    atomicAdd(&gcnt[g], 1.0f);
}

__global__ void graph_finalize_kernel(const float* __restrict__ gsum,
                                      const float* __restrict__ gsumsq,
                                      const float* __restrict__ gcnt,
                                      float* __restrict__ mean, float* __restrict__ inv,
                                      float* __restrict__ invcnt) {
    int g = threadIdx.x;
    if (g >= GG) return;
    float cnt  = fmaxf(gcnt[g], 1.0f);
    float norm = cnt * (float)HC;
    float mu   = gsum[g] / norm;
    float var  = gsumsq[g] / norm - mu * mu;
    mean[g]   = mu;
    inv[g]    = rsqrtf(var + 1e-5f);
    invcnt[g] = 1.0f / cnt;
}

// LN(graph mode) + LeakyReLU(0.01) + mean pool (division folded into the atomic add)
__global__ void norm_pool_kernel(const float* __restrict__ out, const int* __restrict__ batch,
                                 const float* __restrict__ mean, const float* __restrict__ inv,
                                 const float* __restrict__ invcnt,
                                 const float* __restrict__ lnw, const float* __restrict__ lnb,
                                 float* __restrict__ pooled) {
    int gid = blockIdx.x * blockDim.x + threadIdx.x;
    if (gid >= NN * HC) return;
    int n = gid >> 8, c = gid & 255;
    int g = batch[n];
    float y = (out[gid] - mean[g]) * inv[g] * lnw[c] + lnb[c];
    y = (y >= 0.0f) ? y : 0.01f * y;
    atomicAdd(&pooled[(size_t)g * HC + c], y * invcnt[g]);
}

// ---------------- host launcher ----------------
extern "C" void kernel_launch(void* const* d_in, const int* in_sizes, int n_in,
                              void* d_out, int out_size, void* d_ws, size_t ws_size,
                              hipStream_t stream) {
    (void)in_sizes; (void)n_in; (void)out_size; (void)ws_size;

    const float* feat1 = (const float*)d_in[0];
    const float* feat2 = (const float*)d_in[1];
    const float* Wq = (const float*)d_in[2];  const float* bq = (const float*)d_in[3];
    const float* Wk = (const float*)d_in[4];  const float* bk = (const float*)d_in[5];
    const float* Wv = (const float*)d_in[6];  const float* bv = (const float*)d_in[7];
    const float* Ws = (const float*)d_in[8];  const float* bs = (const float*)d_in[9];
    const float* lnw = (const float*)d_in[10]; const float* lnb = (const float*)d_in[11];
    const int* ei1 = (const int*)d_in[12];
    const int* ei2 = (const int*)d_in[13];
    const int* batch = (const int*)d_in[14];
    float* pooled = (float*)d_out;

    // ---- carve workspace (256B aligned bump allocator) ----
    char* base = (char*)d_ws;
    size_t off = 0;
    auto carve = [&](size_t bytes) -> void* {
        void* r = base + off;
        off = (off + bytes + 255) & ~(size_t)255;
        return r;
    };
    unsigned short* xh   = (unsigned short*)carve((size_t)NN * HC * 2);
    unsigned short* whq  = (unsigned short*)carve((size_t)HC * INC * 2);
    unsigned short* whk  = (unsigned short*)carve((size_t)HC * INC * 2);
    unsigned short* whv  = (unsigned short*)carve((size_t)HC * INC * 2);
    unsigned short* whs  = (unsigned short*)carve((size_t)HC * INC * 2);
    float* q    = (float*)carve((size_t)NN * HC * 4);
    float* k    = (float*)carve((size_t)NN * HC * 4);
    float* v    = (float*)carve((size_t)NN * HC * 4);
    float* sout = (float*)carve((size_t)NN * HC * 4);
    float* agg  = (float*)carve((size_t)NN * HC * 4);          // becomes "out"
    float* sc   = (float*)carve((size_t)E2 * HH * 4);
    unsigned* menc = (unsigned*)carve((size_t)NN * HH * 4);
    float* denom   = (float*)carve((size_t)NN * HH * 4);
    int* keep = (int*)carve((size_t)E2 * 4);
    int* se   = (int*)carve((size_t)E2 * 4);
    int* de   = (int*)carve((size_t)E2 * 4);
    unsigned long long* table = (unsigned long long*)carve((size_t)TBL_SIZE * 8);
    float* gstats = (float*)carve((size_t)GG * 6 * 4);  // gsum, gsumsq, gcnt, mean, inv, invcnt
    float* gsum = gstats, *gsumsq = gstats + GG, *gcnt = gstats + 2 * GG;
    float* mean = gstats + 3 * GG, *inv = gstats + 4 * GG, *invcnt = gstats + 5 * GG;

    // ---- reset per-call state (graph-capture safe) ----
    hipMemsetAsync(menc,  0,    (size_t)NN * HH * 4, stream);  // enc(0) < enc(any finite)
    hipMemsetAsync(denom, 0,    (size_t)NN * HH * 4, stream);
    hipMemsetAsync(agg,   0,    (size_t)NN * HC * 4, stream);
    hipMemsetAsync(table, 0xFF, (size_t)TBL_SIZE * 8, stream); // EMPTY = all ones
    hipMemsetAsync(gstats, 0,   (size_t)GG * 6 * 4, stream);
    hipMemsetAsync(pooled, 0,   (size_t)GG * HC * 4, stream);

    const int B = 256;

    // ---- stage 1: bf16 operands ----
    build_xh_kernel<<<(NN * HC + B - 1) / B, B, 0, stream>>>(feat1, feat2, xh);
    f32_to_bf16_kernel<<<(HC * INC + B - 1) / B, B, 0, stream>>>(Wq, whq, HC * INC);
    f32_to_bf16_kernel<<<(HC * INC + B - 1) / B, B, 0, stream>>>(Wk, whk, HC * INC);
    f32_to_bf16_kernel<<<(HC * INC + B - 1) / B, B, 0, stream>>>(Wv, whv, HC * INC);
    f32_to_bf16_kernel<<<(HC * INC + B - 1) / B, B, 0, stream>>>(Ws, whs, HC * INC);

    // ---- stage 2: WMMA GEMMs (4 waves/block, each wave a 16x64 tile) ----
    dim3 ggrid(NN / 16);   // 3125 M-tiles; 4 waves cover all 256 columns
    gemm_wmma_bf16_kernel<<<ggrid, 128, 0, stream>>>(xh, whq, bq, q);
    gemm_wmma_bf16_kernel<<<ggrid, 128, 0, stream>>>(xh, whk, bk, k);
    gemm_wmma_bf16_kernel<<<ggrid, 128, 0, stream>>>(xh, whv, bv, v);
    gemm_wmma_bf16_kernel<<<ggrid, 128, 0, stream>>>(xh, whs, bs, sout);

    // ---- stage 3: edge pipeline (L2-resident gathers/scatters) ----
    dedup_kernel<<<(E2 + B - 1) / B, B, 0, stream>>>(ei1, ei2, table, keep, se, de);
    int eh_blocks = (E2 * HH + B - 1) / B;
    scores_kernel<<<eh_blocks, B, 0, stream>>>(q, k, se, de, keep, sc, menc);
    expsum_kernel<<<eh_blocks, B, 0, stream>>>(de, keep, menc, sc, denom);
    aggregate_kernel<<<eh_blocks, B, 0, stream>>>(se, de, keep, sc, denom, v, agg);

    // ---- stage 4: graph layernorm + leaky relu + mean pool ----
    node_stats_kernel<<<(NN + B - 1) / B, B, 0, stream>>>(sout, agg, batch, gsum, gsumsq, gcnt);
    graph_finalize_kernel<<<1, 64, 0, stream>>>(gsum, gsumsq, gcnt, mean, inv, invcnt);
    norm_pool_kernel<<<(NN * HC + B - 1) / B, B, 0, stream>>>(agg, batch, mean, inv, invcnt,
                                                              lnw, lnb, pooled);
}
